// Head_39247411151485
// MI455X (gfx1250) — matrix operand order repaired and checked by reading, not compiled, not run on hardware
//
#include <hip/hip_runtime.h>
#include <hip/hip_bf16.h>
#include <math.h>

typedef __bf16 bf16_t;
typedef __attribute__((ext_vector_type(16))) __bf16       v16bf;
typedef __attribute__((ext_vector_type(8)))  float        v8f;
typedef __attribute__((ext_vector_type(4)))  unsigned int u32x4;
typedef __attribute__((ext_vector_type(4)))  float        f32x4;

#define B_ 4
#define T_ 2048
#define C_ 1024
#define D_ 128

// ---------------------------------------------------------------------------
// CDNA5 async copy: global -> LDS, 16B per lane, tracked by ASYNCcnt.
// ---------------------------------------------------------------------------
__device__ __forceinline__ void async_copy16(unsigned int lds_addr, const bf16_t* gptr) {
    asm volatile("global_load_async_to_lds_b128 %0, %1, off"
                 :
                 : "v"(lds_addr), "v"((unsigned long long)(uintptr_t)gptr)
                 : "memory");
}

__device__ __forceinline__ void wait_asynccnt0() {
    asm volatile("s_wait_asynccnt 0" ::: "memory");
}

// ---------------------------------------------------------------------------
// Kernel 0: transpose+convert weights: wt[m][d][c] = W_m[c][d] as bf16.
// ---------------------------------------------------------------------------
__global__ void wt_prep(const float* __restrict__ Wq, const float* __restrict__ Wk,
                        const float* __restrict__ Wv, bf16_t* __restrict__ wt) {
    int idx = blockIdx.x * blockDim.x + threadIdx.x;
    if (idx >= 3 * C_ * D_) return;
    int m = idx / (C_ * D_);
    int r = idx % (C_ * D_);
    int c = r / D_;
    int d = r % D_;
    const float* W = (m == 0) ? Wq : ((m == 1) ? Wk : Wv);
    wt[(size_t)m * D_ * C_ + (size_t)d * C_ + c] = (bf16_t)W[(size_t)c * D_ + d];
}

// ---------------------------------------------------------------------------
// Kernel 1: QKV projection. One wave handles one (matrix, 16-row) tile.
//   q,k stored bf16 [B*T][D] row-major (q pre-scaled by D^-1/2).
//   v stored bf16 TRANSPOSED: vT[b][d][t]  (so P@V B-fragments are contiguous)
// ---------------------------------------------------------------------------
__global__ __launch_bounds__(256) void qkv_proj(const float* __restrict__ x,
                                                const bf16_t* __restrict__ wt,
                                                bf16_t* __restrict__ q,
                                                bf16_t* __restrict__ k,
                                                bf16_t* __restrict__ vT) {
    const int lane = threadIdx.x & 31;
    const int wave = threadIdx.x >> 5;
    const int gw   = blockIdx.x * 8 + wave;   // [0, 1536)
    const int m    = gw >> 9;                 // 0=q 1=k 2=v
    const int tile = gw & 511;                // 16-row tile of 8192 rows
    const int row0 = tile * 16;
    const int n16  = lane & 15;
    const int hi   = lane >> 4;

    const bf16_t* wtm  = wt + (size_t)m * D_ * C_;
    const float*  xrow = x + (size_t)(row0 + n16) * C_;

    v8f acc[8];
#pragma unroll
    for (int i = 0; i < 8; ++i) {
#pragma unroll
        for (int j = 0; j < 8; ++j) acc[i][j] = 0.0f;
    }

    for (int kc = 0; kc < C_ / 32; ++kc) {
        // A fragment (16x32 bf16): lane<16 holds K 0..15, lane>=16 K 16..31, contiguous
        const int g0 = kc * 32 + (hi ? 16 : 0);
        f32x4 f0 = *(const f32x4*)(xrow + g0);
        f32x4 f1 = *(const f32x4*)(xrow + g0 + 4);
        f32x4 f2 = *(const f32x4*)(xrow + g0 + 8);
        f32x4 f3 = *(const f32x4*)(xrow + g0 + 12);
        __builtin_prefetch(xrow + g0 + 32, 0, 1);
        v16bf af;
#pragma unroll
        for (int i = 0; i < 4; ++i) {
            af[i]      = (bf16_t)f0[i];
            af[4 + i]  = (bf16_t)f1[i];
            af[8 + i]  = (bf16_t)f2[i];
            af[12 + i] = (bf16_t)f3[i];
        }
#pragma unroll
        for (int nt = 0; nt < 8; ++nt) {
            const int n = nt * 16 + n16;
            // B fragment (32x16): lanes 0-15 K=0..15, lanes 16-31 K=16..31, contiguous
            v16bf bf = *(const v16bf*)(wtm + (size_t)n * C_ + kc * 32 + (hi ? 16 : 0));
            acc[nt] = __builtin_amdgcn_wmma_f32_16x16x32_bf16(
                false, af, false, bf, (short)0, acc[nt], false, false);
        }
    }

    if (m == 0) {
        const float qscale = 0.08838834764831845f;  // 1/sqrt(D)
#pragma unroll
        for (int nt = 0; nt < 8; ++nt)
#pragma unroll
            for (int r = 0; r < 8; ++r) acc[nt][r] *= qscale;
    }

    // C/D layout: VGPR r <-> row (r + 8*hi), column = nt*16 + n16
    if (m < 2) {
        bf16_t* outp = (m == 0) ? q : k;
#pragma unroll
        for (int nt = 0; nt < 8; ++nt)
#pragma unroll
            for (int r = 0; r < 8; ++r)
                outp[(size_t)(row0 + hi * 8 + r) * D_ + nt * 16 + n16] = (bf16_t)acc[nt][r];
    } else {
        const int b     = row0 / T_;
        const int tloc0 = row0 - b * T_ + hi * 8;
#pragma unroll
        for (int nt = 0; nt < 8; ++nt) {
            const int d = nt * 16 + n16;
#pragma unroll
            for (int r = 0; r < 8; ++r)
                vT[((size_t)b * D_ + d) * T_ + tloc0 + r] = (bf16_t)acc[nt][r];
        }
    }
}

// ---------------------------------------------------------------------------
// Kernel 2: fused causal flash attention.
//   Block = 8 waves = 128 query rows.  Key chunks of 32 are staged into LDS
//   once per block with CDNA5 async-to-LDS copies (double-buffered, ASYNCcnt),
//   then all waves read their WMMA fragments from LDS.
// ---------------------------------------------------------------------------
__global__ __launch_bounds__(256) void attn(const bf16_t* __restrict__ q,
                                            const bf16_t* __restrict__ k,
                                            const bf16_t* __restrict__ vT,
                                            float* __restrict__ out) {
    __shared__ __align__(32) bf16_t Kbuf[2][32][128];   // 2 x 8KB
    __shared__ __align__(32) bf16_t Vbuf[2][128][32];   // 2 x 8KB
    __shared__ __align__(32) bf16_t plds[8][16 * 32];   // per-wave P tile, 8KB

    const int tid  = threadIdx.x;
    const int lane = tid & 31;
    const int wave = tid >> 5;
    const int b    = blockIdx.y;
    const int t0   = blockIdx.x * 128 + wave * 16;  // query tile start (within batch)
    const int n16  = lane & 15;
    const int hi   = lane >> 4;

    const bf16_t* qb = q + (size_t)b * T_ * D_;
    const bf16_t* kb = k + (size_t)b * T_ * D_;
    const bf16_t* vb = vT + (size_t)b * D_ * T_;

    // ---- staging lambda: copy K chunk (32x128) + V chunk (128x32) to LDS ----
    auto stage = [&](int j0, int bufi) {
        // K: 8KB contiguous, 256 threads x 2 x 16B
        {
            const bf16_t* src = kb + (size_t)j0 * D_ + tid * 8;
            unsigned int dst = (unsigned int)(uintptr_t)(&Kbuf[bufi][0][0]) + tid * 16;
            async_copy16(dst, src);
            async_copy16(dst + 4096, src + 2048);
        }
        // V: 128 rows x 64B (global stride T_), 2 threads per row x 2 x 16B
        {
            const int row = tid >> 1;
            const int off = (tid & 1) * 16;  // halves
            const bf16_t* src = vb + (size_t)row * T_ + j0 + off;
            unsigned int dst =
                (unsigned int)(uintptr_t)(&Vbuf[bufi][0][0]) + row * 64 + off * 2;
            async_copy16(dst, src);
            async_copy16(dst + 16, src + 8);
        }
    };

    // ---- Q fragments for all of D=128 (4 x K=32), loaded once ----
    v16bf qa[4];
    {
        const bf16_t* qrow = qb + (size_t)(t0 + n16) * D_;
#pragma unroll
        for (int c = 0; c < 4; ++c) {
            const int g0 = c * 32 + (hi ? 16 : 0);
            u32x4* dst = (u32x4*)&qa[c];
            dst[0] = *(const u32x4*)(qrow + g0);
            dst[1] = *(const u32x4*)(qrow + g0 + 8);
        }
    }

    v8f O[8];
#pragma unroll
    for (int i = 0; i < 8; ++i) {
#pragma unroll
        for (int j = 0; j < 8; ++j) O[i][j] = 0.0f;
    }
    float mrow[8], lrow[8];
#pragma unroll
    for (int r = 0; r < 8; ++r) { mrow[r] = -__builtin_inff(); lrow[r] = 0.0f; }

    bf16_t* myp = &plds[wave][0];
    const int myChunks  = (t0 + 15) / 32 + 1;    // this wave's causal bound
    const int maxChunks = blockIdx.x * 4 + 4;    // block-wide bound (uniform)

    stage(0, 0);  // preload chunk 0

    for (int jc = 0; jc < maxChunks; ++jc) {
        wait_asynccnt0();   // my async loads for buffer jc&1 have landed
        __syncthreads();    // everyone's landed; prior buffer fully consumed

        if (jc + 1 < maxChunks) stage((jc + 1) * 32, (jc + 1) & 1);

        if (jc < myChunks) {  // wave-uniform; EXEC stays all-ones inside
            const int j0  = jc * 32;
            const int buf = jc & 1;

            // ---- S = Q @ K^T for two 16-key sub-tiles (fragments from LDS) ----
            v8f s[2];
#pragma unroll
            for (int sf = 0; sf < 2; ++sf) {
#pragma unroll
                for (int j = 0; j < 8; ++j) s[sf][j] = 0.0f;
                const bf16_t* krow = &Kbuf[buf][sf * 16 + n16][hi ? 16 : 0];
#pragma unroll
                for (int c = 0; c < 4; ++c) {
                    v16bf kf = *(const v16bf*)(krow + c * 32);
                    s[sf] = __builtin_amdgcn_wmma_f32_16x16x32_bf16(
                        false, qa[c], false, kf, (short)0, s[sf], false, false);
                }
            }

            // ---- causal mask ----
            if (j0 + 31 > t0) {
#pragma unroll
                for (int sf = 0; sf < 2; ++sf)
#pragma unroll
                    for (int r = 0; r < 8; ++r) {
                        const int trow = t0 + r + hi * 8;
                        const int jcol = j0 + sf * 16 + n16;
                        if (jcol > trow) s[sf][r] = -__builtin_inff();
                    }
            }

            // ---- online softmax across the 16 lanes holding each row ----
#pragma unroll
            for (int r = 0; r < 8; ++r) {
                float v = fmaxf(s[0][r], s[1][r]);
                v = fmaxf(v, __shfl_xor(v, 1));
                v = fmaxf(v, __shfl_xor(v, 2));
                v = fmaxf(v, __shfl_xor(v, 4));
                v = fmaxf(v, __shfl_xor(v, 8));
                const float mnew  = fmaxf(mrow[r], v);
                const float alpha = __expf(mrow[r] - mnew);
                mrow[r] = mnew;
                const float p0 = __expf(s[0][r] - mnew);
                const float p1 = __expf(s[1][r] - mnew);
                s[0][r] = p0;
                s[1][r] = p1;
                float rs = p0 + p1;
                rs += __shfl_xor(rs, 1);
                rs += __shfl_xor(rs, 2);
                rs += __shfl_xor(rs, 4);
                rs += __shfl_xor(rs, 8);
                lrow[r] = lrow[r] * alpha + rs;
#pragma unroll
                for (int nt = 0; nt < 8; ++nt) O[nt][r] *= alpha;
            }

            // ---- re-layout P (C layout -> A layout) through per-wave LDS ----
#pragma unroll
            for (int sf = 0; sf < 2; ++sf)
#pragma unroll
                for (int r = 0; r < 8; ++r)
                    myp[(r + hi * 8) * 32 + sf * 16 + n16] = (bf16_t)s[sf][r];
            asm volatile("s_wait_dscnt 0" ::: "memory");  // wave-local cross-lane

            v16bf pf;
            {
                const bf16_t* prow = myp + n16 * 32 + (hi ? 16 : 0);
                u32x4* dst = (u32x4*)&pf;
                dst[0] = *(const u32x4*)(prow);
                dst[1] = *(const u32x4*)(prow + 8);
            }

            // ---- O += P @ V (V fragments contiguous from LDS) ----
#pragma unroll
            for (int nt = 0; nt < 8; ++nt) {
                const bf16_t* vrow = &Vbuf[buf][nt * 16 + n16][hi ? 16 : 0];
                v16bf vf = *(const v16bf*)vrow;
                O[nt] = __builtin_amdgcn_wmma_f32_16x16x32_bf16(
                    false, pf, false, vf, (short)0, O[nt], false, false);
            }
        }
    }

    // ---- epilogue: out = O / l ----
    float* ob = out + ((size_t)b * T_ + t0) * D_;
#pragma unroll
    for (int r = 0; r < 8; ++r) {
        const float inv = 1.0f / lrow[r];
#pragma unroll
        for (int nt = 0; nt < 8; ++nt)
            ob[(size_t)(r + hi * 8) * D_ + nt * 16 + n16] = O[nt][r] * inv;
    }
}

// ---------------------------------------------------------------------------
extern "C" void kernel_launch(void* const* d_in, const int* in_sizes, int n_in,
                              void* d_out, int out_size, void* d_ws, size_t ws_size,
                              hipStream_t stream) {
    const float* x  = (const float*)d_in[0];
    const float* Wq = (const float*)d_in[1];
    const float* Wk = (const float*)d_in[2];
    const float* Wv = (const float*)d_in[3];
    float* out = (float*)d_out;

    char* ws = (char*)d_ws;
    bf16_t* wt = (bf16_t*)ws;                                  // 3*128*1024*2B = 768 KB
    bf16_t* qb = (bf16_t*)(ws + (size_t)786432);               // 8192*128*2B   = 2 MB
    bf16_t* kb = (bf16_t*)(ws + (size_t)786432 + 2097152);     // 2 MB
    bf16_t* vT = (bf16_t*)(ws + (size_t)786432 + 2 * 2097152); // 4*128*2048*2B = 2 MB

    hipLaunchKernelGGL(wt_prep, dim3((3 * C_ * D_ + 255) / 256), dim3(256), 0, stream,
                       Wq, Wk, Wv, wt);
    hipLaunchKernelGGL(qkv_proj, dim3(192), dim3(256), 0, stream, x, wt, qb, kb, vT);
    hipLaunchKernelGGL(attn, dim3(16, 4), dim3(256), 0, stream, qb, kb, vT, out);
}
